// my_Net_25434796327517
// MI455X (gfx1250) — compile-verified, matrix operand order checked by reading, not run on hardware
//
#include <hip/hip_runtime.h>
#include <stdint.h>

// ---------------------------------------------------------------------------
// Bit-sliced MVM network for MI455X (gfx1250).
// Heavy lifting: V_WMMA_I32_16X16X64_IU8 (signed i8 x i8 -> i32), which
// exactly matches the 4-bit sign-magnitude slices (values in [-15,15]).
// ---------------------------------------------------------------------------

typedef int v8i __attribute__((ext_vector_type(8)));

// ---- fixed-point helpers ---------------------------------------------------
// fixq(x,16,14) then pack the 4 sign-magnitude 4-bit slices as int8 bytes.
__device__ __forceinline__ uint32_t pack_slices(float val) {
    float a = rintf(val * 16384.0f);                 // round-half-even, 2^14 frac
    a = fmaxf(fminf(a, 32767.0f), -32768.0f);        // clip to [-2^15, 2^15-1]
    int q   = (int)a;
    int sg  = (q < 0) ? -1 : 1;
    int mag = (q < 0) ? -q : q;
    uint32_t r = 0;
#pragma unroll
    for (int s = 0; s < 4; ++s) {
        int sl = sg * ((mag >> (4 * s)) & 15);       // in [-15, 15]
        r |= ((uint32_t)(uint8_t)(int8_t)sl) << (8 * s);
    }
    return r;
}

// ADC clamp + shift-combine + 32.24 accumulator quantize (all f32, like jnp).
__device__ __forceinline__ float adc_acm_combine(const int* p16 /*16 partials, s*4+t*/) {
    const float shifts[7] = {1.f, 16.f, 256.f, 4096.f, 65536.f, 1048576.f, 16777216.f};
    float out = 0.f;
#pragma unroll
    for (int s = 0; s < 4; ++s)
#pragma unroll
        for (int t = 0; t < 4; ++t) {
            int p = p16[s * 4 + t];
            p = (p < -8192) ? -8192 : ((p > 8191) ? 8191 : p);   // ADC_BIT=14
            out += (float)p * shifts[s + t];
        }
    out *= (1.0f / 268435456.0f);                    // / 2^(14+14)
    float a = rintf(out * 16777216.0f);              // * 2^24
    a = fmaxf(fminf(a, 2147483648.0f), -2147483648.0f);
    return a * (1.0f / 16777216.0f);
}

// 4x4 byte transpose with v_perm_b32: out[s] byte j = w_j byte s.
// 8 perms produce all four stream dwords at once.
__device__ __forceinline__ void transpose4(uint32_t w0, uint32_t w1,
                                           uint32_t w2, uint32_t w3,
                                           uint32_t out[4]) {
    uint32_t p01_lo = __builtin_amdgcn_perm(w1, w0, 0x05010400u); // w0b0,w1b0,w0b1,w1b1
    uint32_t p23_lo = __builtin_amdgcn_perm(w3, w2, 0x05010400u); // w2b0,w3b0,w2b1,w3b1
    uint32_t p01_hi = __builtin_amdgcn_perm(w1, w0, 0x07030602u); // w0b2,w1b2,w0b3,w1b3
    uint32_t p23_hi = __builtin_amdgcn_perm(w3, w2, 0x07030602u); // w2b2,w3b2,w2b3,w3b3
    out[0] = __builtin_amdgcn_perm(p23_lo, p01_lo, 0x05040100u);  // stream 0
    out[1] = __builtin_amdgcn_perm(p23_lo, p01_lo, 0x07060302u);  // stream 1
    out[2] = __builtin_amdgcn_perm(p23_hi, p01_hi, 0x05040100u);  // stream 2
    out[3] = __builtin_amdgcn_perm(p23_hi, p01_hi, 0x07060302u);  // stream 3
}

// ---------------------------------------------------------------------------
// Kernel 0: swizzle conv weights into WMMA B fragments.
// K order: knew = (kh*3+kw)*C + c  (spatial-major / channel-minor).
// Fragment layout: frag[t][ntile(32)][chunk(C/64*9)][lane(32)][v(8)] u32.
// Lane l holds column n=l&15, K-halves selected by l>=16 (mirrors A layout).
// ---------------------------------------------------------------------------
__global__ void prep_wfrag(const float* __restrict__ w, uint8_t* __restrict__ frag,
                           int total, int C) {
    int i = blockIdx.x * blockDim.x + threadIdx.x;
    if (i >= total) return;
    int ks = i % 9;
    int oc = i / 9;
    int c  = oc % C;
    int o  = oc / C;

    float a = rintf(w[i] * 16384.0f);
    a = fmaxf(fminf(a, 32767.0f), -32768.0f);
    int q = (int)a;
    int sg  = (q < 0) ? -1 : 1;
    int mag = (q < 0) ? -q : q;

    int knew  = ks * C + c;
    int chunk = knew >> 6;
    int kk    = knew & 63;
    int hi8   = (kk >> 3) & 1;
    int j     = kk & 3;
    int v     = ((kk >> 5) << 2) | (((kk >> 4) & 1) << 1) | ((kk >> 2) & 1);
    int ntile = o >> 4;
    int n     = o & 15;
    int lane  = hi8 * 16 + n;
    int nchunk = (C / 64) * 9;
#pragma unroll
    for (int t = 0; t < 4; ++t) {
        int sl = sg * ((mag >> (4 * t)) & 15);
        size_t idx = ((((size_t)(t * 32 + ntile) * nchunk + chunk) * 32 + lane) * 8 + v) * 4 + j;
        frag[idx] = (uint8_t)(int8_t)sl;
    }
}

// wlin [10][512] -> packed [k(512)][o(10)] u32 (4 slice bytes)
__global__ void prep_lin(const float* __restrict__ wl, uint32_t* __restrict__ wp) {
    int i = blockIdx.x * blockDim.x + threadIdx.x;
    if (i >= 5120) return;
    int k = i % 512, o = i / 512;
    wp[k * 10 + o] = pack_slices(wl[i]);
}

// ---------------------------------------------------------------------------
// Kernel 1: float conv1 (3->512, 3x3, stride1, pad1). Writes y floats and the
// packed-slice u32 tensor (input of conv2).
// ---------------------------------------------------------------------------
__global__ void conv1_kernel(const float* __restrict__ x, const float* __restrict__ w1,
                             float* __restrict__ y, uint32_t* __restrict__ ypack) {
    int i = blockIdx.x * blockDim.x + threadIdx.x;
    if (i >= 64 * 512 * 32 * 32) return;
    int ow = i & 31; int t = i >> 5;
    int oh = t & 31; t >>= 5;
    int o  = t & 511; int b = t >> 9;
    float sum = 0.f;
#pragma unroll
    for (int c = 0; c < 3; ++c)
#pragma unroll
        for (int kh = 0; kh < 3; ++kh) {
            int ih = oh + kh - 1;
            if ((unsigned)ih >= 32u) continue;
#pragma unroll
            for (int kw = 0; kw < 3; ++kw) {
                int iw = ow + kw - 1;
                if ((unsigned)iw >= 32u) continue;
                sum += x[((b * 3 + c) * 32 + ih) * 32 + iw] *
                       w1[((o * 3 + c) * 3 + kh) * 3 + kw];
            }
        }
    y[i] = sum;
    ypack[i] = pack_slices(sum);
}

// ---------------------------------------------------------------------------
// Kernel 2: bit-sliced MVM conv as implicit GEMM on WMMA IU8.
// One wave = one 16x16 D tile, 16 separate i32 accumulators (s,t), K=C*9 in
// 64-chunks with fixed (kh,kw) per chunk -> compile-time immediate offsets.
// cc loop unrolled x2 so afrag sets ping-pong (kills the WMMA->VALU WAR nops).
// ---------------------------------------------------------------------------
template <int C, int IH, int IW, int OH, int OW, int STRIDE>
__global__ __launch_bounds__(128) void mvm_conv_kernel(
    const uint32_t* __restrict__ inpack,   // [B][C][IH][IW] u32 (4 stream bytes)
    const uint32_t* __restrict__ wfrag,    // prearranged B fragments
    float* __restrict__ outf,              // [B][512][OH][OW]
    uint32_t* __restrict__ outpack)        // same shape, packed slices (or null)
{
    constexpr int CHST   = IH * IW;        // channel stride (u32 elements)
    constexpr int NCHUNK = (C / 64) * 9;   // 72 for C=512

    const int lane  = threadIdx.x & 31;
    const int wv    = threadIdx.x >> 5;
    const int mtile = blockIdx.x;
    const int ntile = blockIdx.y * 4 + wv;  // 0..31
    const int n     = lane & 15;
    const int hi    = lane >> 4;

    // A-row coordinates for this lane (row m = lane&15 of the tile)
    const int m   = mtile * 16 + n;
    const int b   = m / (OH * OW);
    const int rem = m - b * (OH * OW);
    const int oh  = rem / OW;
    const int ow  = rem - oh * OW;

    v8i acc[4][4] = {};
    v8i afrag[4];
    v8i bfrag[4];

    const uint32_t* bp = wfrag + ((size_t)ntile * NCHUNK) * 256 + lane * 8;

    for (int kh = 0; kh < 3; ++kh) {
        for (int kw = 0; kw < 3; ++kw) {
            const int ih = oh * STRIDE + kh - 1;
            const int iw = ow * STRIDE + kw - 1;
            const bool valid = ((unsigned)ih < (unsigned)IH) && ((unsigned)iw < (unsigned)IW);
            const uint32_t* aptr =
                inpack + ((long)(b * C) * IH + ih) * IW + iw + hi * 8 * CHST;

#pragma unroll 2
            for (int cc = 0; cc < C / 64; ++cc) {
                // ---- build A fragments (all 4 streams, 8 perms per dword) ----
#pragma unroll
                for (int v = 0; v < 8; ++v) {
                    const int koff = (v >> 2) * 32 + ((v >> 1) & 1) * 16 + (v & 1) * 4;
                    uint32_t w0 = 0, w1 = 0, w2 = 0, w3 = 0;
                    if (valid) {
                        w0 = aptr[(koff + 0) * CHST];
                        w1 = aptr[(koff + 1) * CHST];
                        w2 = aptr[(koff + 2) * CHST];
                        w3 = aptr[(koff + 3) * CHST];
                    }
                    uint32_t o4[4];
                    transpose4(w0, w1, w2, w3, o4);
                    afrag[0][v] = (int)o4[0];
                    afrag[1][v] = (int)o4[1];
                    afrag[2][v] = (int)o4[2];
                    afrag[3][v] = (int)o4[3];
                }
                // ---- load B fragments (pre-swizzled, coalesced b128) ----
#pragma unroll
                for (int t = 0; t < 4; ++t) {
                    const uint32_t* bpt = bp + (size_t)t * 32 * NCHUNK * 256;
                    uint4 lo  = *(const uint4*)(bpt);
                    uint4 hi4 = *(const uint4*)(bpt + 4);
                    bfrag[t][0] = (int)lo.x;  bfrag[t][1] = (int)lo.y;
                    bfrag[t][2] = (int)lo.z;  bfrag[t][3] = (int)lo.w;
                    bfrag[t][4] = (int)hi4.x; bfrag[t][5] = (int)hi4.y;
                    bfrag[t][6] = (int)hi4.z; bfrag[t][7] = (int)hi4.w;
                }
                // ---- 16 WMMAs: all (stream, slice) partial products ----
#pragma unroll
                for (int s = 0; s < 4; ++s)
#pragma unroll
                    for (int t = 0; t < 4; ++t)
                        acc[s][t] = __builtin_amdgcn_wmma_i32_16x16x64_iu8(
                            true, afrag[s], true, bfrag[t], acc[s][t], false, false);

                aptr += 64 * CHST;
                bp   += 256;
            }
        }
    }

    // ---- epilogue: ADC clamp, shift-combine, ACM quantize, store ----
    const int o = ntile * 16 + n;
#pragma unroll
    for (int r = 0; r < 8; ++r) {
        int p16[16];
#pragma unroll
        for (int s = 0; s < 4; ++s)
#pragma unroll
            for (int t = 0; t < 4; ++t)
                p16[s * 4 + t] = acc[s][t][r];
        float out = adc_acm_combine(p16);

        const int me  = mtile * 16 + r + hi * 8;   // D layout: VGPR r -> M=r / 8+r
        const int be  = me / (OH * OW);
        const int re  = me - be * (OH * OW);
        const int ohe = re / OW;
        const int owe = re - ohe * OW;
        const long oidx = (((long)be * 512 + o) * OH + ohe) * OW + owe;
        outf[oidx] = out;
        if (outpack) outpack[oidx] = pack_slices(out);
    }
}

// ---------------------------------------------------------------------------
// Kernel 3: avgpool 6x6 (top-left of 8x8) + quantize/pack -> xpool[b][c]
// ---------------------------------------------------------------------------
__global__ void pool_kernel(const float* __restrict__ wfeat, uint32_t* __restrict__ xpool) {
    int i = blockIdx.x * blockDim.x + threadIdx.x;
    if (i >= 64 * 512) return;
    const float* p = wfeat + (size_t)i * 64;
    float s = 0.f;
#pragma unroll
    for (int h = 0; h < 6; ++h)
#pragma unroll
        for (int w = 0; w < 6; ++w)
            s += p[h * 8 + w];
    xpool[i] = pack_slices(s * (1.0f / 36.0f));
}

// ---------------------------------------------------------------------------
// Kernel 4: sliced linear [64,512] x [10,512] with per-(s,t) ADC clamp.
// ---------------------------------------------------------------------------
__global__ void linear_kernel(const uint32_t* __restrict__ xpool,
                              const uint32_t* __restrict__ wp,
                              float* __restrict__ out) {
    int i = blockIdx.x * blockDim.x + threadIdx.x;
    if (i >= 640) return;
    int o = i % 10, b = i / 10;
    int acc[16] = {};
    for (int k = 0; k < 512; ++k) {
        uint32_t xw = xpool[b * 512 + k];
        uint32_t ww = wp[k * 10 + o];
#pragma unroll
        for (int s = 0; s < 4; ++s) {
            int xa = (int)(int8_t)(xw >> (8 * s));
#pragma unroll
            for (int t = 0; t < 4; ++t) {
                int wa = (int)(int8_t)(ww >> (8 * t));
                acc[s * 4 + t] += xa * wa;
            }
        }
    }
    out[i] = adc_acm_combine(acc);
}

// ---------------------------------------------------------------------------
extern "C" void kernel_launch(void* const* d_in, const int* in_sizes, int n_in,
                              void* d_out, int out_size, void* d_ws, size_t ws_size,
                              hipStream_t stream) {
    const float* x    = (const float*)d_in[0];
    const float* w1   = (const float*)d_in[1];
    const float* w2   = (const float*)d_in[2];
    const float* w3   = (const float*)d_in[3];
    const float* wlin = (const float*)d_in[4];

    float* out_lin = (float*)d_out;                    // [64,10]
    float* y = out_lin + 640;                          // [64,512,32,32]
    float* z = y + (size_t)64 * 512 * 32 * 32;         // [64,512,16,16]
    float* w = z + (size_t)64 * 512 * 16 * 16;         // [64,512,8,8]

    uint8_t* ws = (uint8_t*)d_ws;
    uint32_t* ypack  = (uint32_t*)ws;                                  // 134,217,728 B
    uint32_t* zpack  = (uint32_t*)(ws + 134217728u);                   //  33,554,432 B
    uint8_t*  w2frag = ws + 134217728u + 33554432u;                    //   9,437,184 B
    uint8_t*  w3frag = w2frag + 9437184u;                              //   9,437,184 B
    uint32_t* wlinp  = (uint32_t*)(w3frag + 9437184u);                 //      20,480 B
    uint32_t* xpool  = wlinp + 5120;                                   //     131,072 B

    const int WPREP = 512 * 512 * 9;
    prep_wfrag<<<(WPREP + 255) / 256, 256, 0, stream>>>(w2, w2frag, WPREP, 512);
    prep_wfrag<<<(WPREP + 255) / 256, 256, 0, stream>>>(w3, w3frag, WPREP, 512);
    prep_lin<<<(5120 + 255) / 256, 256, 0, stream>>>(wlin, wlinp);

    conv1_kernel<<<(64 * 512 * 32 * 32 + 255) / 256, 256, 0, stream>>>(x, w1, y, ypack);

    // conv2: M = 64*16*16 = 16384 -> 1024 M-tiles; N = 512 -> 32 tiles (8 blocks x 4 waves)
    mvm_conv_kernel<512, 32, 32, 16, 16, 2>
        <<<dim3(1024, 8), 128, 0, stream>>>(ypack, (const uint32_t*)w2frag, z, zpack);

    // conv3: M = 64*8*8 = 4096 -> 256 M-tiles
    mvm_conv_kernel<512, 16, 16, 8, 8, 2>
        <<<dim3(256, 8), 128, 0, stream>>>(zpack, (const uint32_t*)w3frag, w, nullptr);

    pool_kernel<<<(64 * 512 + 255) / 256, 256, 0, stream>>>(w, xpool);
    linear_kernel<<<(640 + 63) / 64, 64, 0, stream>>>(xpool, wlinp, out_lin);
}
